// ResidueGraphModel_56453050138688
// MI455X (gfx1250) — matrix operand order repaired and checked by reading, not compiled, not run on hardware
//
#include <hip/hip_runtime.h>
#include <hip/hip_bf16.h>

#define N_NODES 100000
#define N_EDGES 1600000
#define HID 128
#define IN_DIM 512
#define NP1 (N_NODES + 1)

typedef __attribute__((ext_vector_type(16))) __bf16          v16bf;
typedef __attribute__((ext_vector_type(16))) unsigned short  v16us;
typedef __attribute__((ext_vector_type(8)))  float           v8f;

__device__ __forceinline__ v8f wmma_bf16(v16bf a, v16bf b, v8f c) {
  // D = A(16x32 bf16) * B(32x16 bf16) + C(16x16 f32)
  return __builtin_amdgcn_wmma_f32_16x16x32_bf16(
      /*neg_a=*/false, a, /*neg_b=*/false, b,
      /*c_mod=*/(short)0, c, /*reuse_a=*/false, /*reuse_b=*/false);
}

// Native f32 -> bf16 converts (single v_cvt per element, ideally packed).
__device__ __forceinline__ v16bf pack_a16(const float* h) {
  v16bf a;
#pragma unroll
  for (int j = 0; j < 16; ++j) a[j] = (__bf16)h[j];
  return a;
}

// ---------------------------------------------------------------------------
// Pack a [K,128] fp32 weight matrix into per-tile, per-lane bf16 B layout.
// Tile (kt,nt): 512 bf16 values; lane l element j  <->  W[kt*32 + 8*(l>>4) + j + (j>=8 ? 8 : 0)][nt*16 + (l&15)]
// gridDim.x = (K/32)*8 tiles, blockDim.x = 32
__global__ void __launch_bounds__(32) pack_w_kernel(const float* __restrict__ W,
                                                    unsigned short* __restrict__ out) {
  int tile = blockIdx.x;
  int lane = threadIdx.x;
  int kt = tile >> 3, nt = tile & 7;
  int n = nt * 16 + (lane & 15);
  int kbase = kt * 32 + ((lane >> 4) << 3);
  unsigned short* o = out + (size_t)tile * 512 + (size_t)lane * 16;
#pragma unroll
  for (int j = 0; j < 16; ++j) {
    __bf16 b = (__bf16)W[(size_t)(kbase + j + (j >= 8 ? 8 : 0)) * HID + n];
    o[j] = __builtin_bit_cast(unsigned short, b);
  }
}

// ---------------------------------------------------------------------------
// x[0:N] = pep @ Wproj + b    (one wave per 16-row tile, 8 N-tiles, K=512)
__global__ void __launch_bounds__(128) proj_gemm_kernel(const float* __restrict__ A,
                                                        const unsigned short* __restrict__ Wp,
                                                        const float* __restrict__ bias,
                                                        float* __restrict__ x) {
  int wave = blockIdx.x * 4 + (threadIdx.x >> 5);
  if (wave >= N_NODES / 16) return;
  int lane = threadIdx.x & 31;
  int lr = lane & 15, hi = lane >> 4;
  int row = wave * 16 + lr;

  v8f zero = {0.f, 0.f, 0.f, 0.f, 0.f, 0.f, 0.f, 0.f};
  v8f acc[8];
#pragma unroll
  for (int nt = 0; nt < 8; ++nt) acc[nt] = zero;

  const float* arow = A + (size_t)row * IN_DIM;
#pragma unroll 1
  for (int kt = 0; kt < IN_DIM / 32; ++kt) {
    const float* p0 = arow + kt * 32 + hi * 8;        // K = 8*hi + j      (j<8)
    const float* p1 = p0 + 16;                        // K = 16 + 8*hi + j (j>=8)
    float4 f0 = *(const float4*)(p0);
    float4 f1 = *(const float4*)(p0 + 4);
    float4 f2 = *(const float4*)(p1);
    float4 f3 = *(const float4*)(p1 + 4);
    float t[16] = {f0.x, f0.y, f0.z, f0.w, f1.x, f1.y, f1.z, f1.w,
                   f2.x, f2.y, f2.z, f2.w, f3.x, f3.y, f3.z, f3.w};
    v16bf a = pack_a16(t);
#pragma unroll
    for (int nt = 0; nt < 8; ++nt) {
      v16us bu = *(const v16us*)(Wp + ((size_t)(kt * 8 + nt)) * 512 + (size_t)lane * 16);
      acc[nt] = wmma_bf16(a, __builtin_bit_cast(v16bf, bu), acc[nt]);
    }
  }
#pragma unroll
  for (int nt = 0; nt < 8; ++nt) {
    int ocol = nt * 16 + lr;
    float bv = bias[ocol];
#pragma unroll
    for (int r = 0; r < 8; ++r) {
      int orow = wave * 16 + hi * 8 + r;
      x[(size_t)orow * HID + ocol] = acc[nt][r] + bv;
    }
  }
}

__global__ void __launch_bounds__(128) cls_init_kernel(const float* __restrict__ cls,
                                                       float* __restrict__ x) {
  int i = threadIdx.x;
  x[(size_t)N_NODES * HID + i] = cls[i];
}

__global__ void __launch_bounds__(256) zero_kernel(float4* __restrict__ p, int n4) {
  int i = blockIdx.x * blockDim.x + threadIdx.x;
  if (i < n4) p[i] = make_float4(0.f, 0.f, 0.f, 0.f);
}

// ---------------------------------------------------------------------------
// agg[dst] += relu(x[src] + e)  over 1.6M real edges + 100k virtual edges to CLS
__global__ void __launch_bounds__(256) edge_msg_kernel(const float* __restrict__ x,
                                                       const int* __restrict__ ei,
                                                       const int* __restrict__ ea,
                                                       const float* __restrict__ emb,
                                                       const float* __restrict__ cls_edge,
                                                       float* __restrict__ agg) {
  long long t = (long long)blockIdx.x * 256 + threadIdx.x;
  const long long total = (long long)(N_EDGES + N_NODES) * (HID / 4);
  if (t >= total) return;
  int e = (int)(t >> 5);
  int c = ((int)t & 31) * 4;
  int src, dst;
  const float* ev;
  if (e < N_EDGES) {
    src = ei[e];
    dst = ei[N_EDGES + e];
    ev = emb + (size_t)ea[e] * HID + c;
  } else {
    src = e - N_EDGES;
    dst = N_NODES;
    ev = cls_edge + c;
  }
  float4 xv = *(const float4*)(x + (size_t)src * HID + c);
  float4 em = *(const float4*)ev;
  float m0 = fmaxf(xv.x + em.x, 0.f);
  float m1 = fmaxf(xv.y + em.y, 0.f);
  float m2 = fmaxf(xv.z + em.z, 0.f);
  float m3 = fmaxf(xv.w + em.w, 0.f);
  float* ap = agg + (size_t)dst * HID + c;
  __hip_atomic_fetch_add(ap + 0, m0, __ATOMIC_RELAXED, __HIP_MEMORY_SCOPE_AGENT);
  __hip_atomic_fetch_add(ap + 1, m1, __ATOMIC_RELAXED, __HIP_MEMORY_SCOPE_AGENT);
  __hip_atomic_fetch_add(ap + 2, m2, __ATOMIC_RELAXED, __HIP_MEMORY_SCOPE_AGENT);
  __hip_atomic_fetch_add(ap + 3, m3, __ATOMIC_RELAXED, __HIP_MEMORY_SCOPE_AGENT);
}

// ---------------------------------------------------------------------------
// x = relu((x+agg)@W1 + b1) @ W2 + b2 + x   (fused, in place; one wave / 16 rows)
__global__ void __launch_bounds__(128) mlp_kernel(const float* __restrict__ x,
                                                  const float* __restrict__ agg,
                                                  const unsigned short* __restrict__ W1p,
                                                  const float* __restrict__ b1,
                                                  const unsigned short* __restrict__ W2p,
                                                  const float* __restrict__ b2,
                                                  float* __restrict__ xout) {
  __shared__ float lds[4][16][132];
  int widx = threadIdx.x >> 5;
  int wave = blockIdx.x * 4 + widx;
  int lane = threadIdx.x & 31;
  int lr = lane & 15, hi = lane >> 4;
  bool active = (wave * 16) < NP1;
  int row = wave * 16 + lr;
  if (row >= NP1) row = NP1 - 1;   // only the last tile clamps, to its own row

  v8f zero = {0.f, 0.f, 0.f, 0.f, 0.f, 0.f, 0.f, 0.f};

  if (active) {
    v8f acc[8];
#pragma unroll
    for (int nt = 0; nt < 8; ++nt) acc[nt] = zero;
    const float* xp = x + (size_t)row * HID;
    const float* gp = agg + (size_t)row * HID;
#pragma unroll
    for (int kt = 0; kt < HID / 32; ++kt) {
      int b0 = kt * 32 + hi * 8;
      float t[16];
#pragma unroll
      for (int j = 0; j < 8; ++j) t[j] = xp[b0 + j] + gp[b0 + j];
#pragma unroll
      for (int j = 0; j < 8; ++j) t[8 + j] = xp[b0 + 16 + j] + gp[b0 + 16 + j];
      v16bf a = pack_a16(t);
#pragma unroll
      for (int nt = 0; nt < 8; ++nt) {
        v16us bu = *(const v16us*)(W1p + ((size_t)(kt * 8 + nt)) * 512 + (size_t)lane * 16);
        acc[nt] = wmma_bf16(a, __builtin_bit_cast(v16bf, bu), acc[nt]);
      }
    }
    // bias + relu -> LDS (transpose C layout -> row-major for reuse as A)
#pragma unroll
    for (int nt = 0; nt < 8; ++nt) {
      int col = nt * 16 + lr;
      float bv = b1[col];
#pragma unroll
      for (int r = 0; r < 8; ++r) {
        float v = acc[nt][r] + bv;
        lds[widx][hi * 8 + r][col] = v > 0.f ? v : 0.f;
      }
    }
  }
  __syncthreads();
  if (active) {
    v8f acc2[8];
#pragma unroll
    for (int nt = 0; nt < 8; ++nt) acc2[nt] = zero;
#pragma unroll
    for (int kt = 0; kt < HID / 32; ++kt) {
      int b0 = kt * 32 + hi * 8;
      float t[16];
#pragma unroll
      for (int j = 0; j < 8; ++j) t[j] = lds[widx][lr][b0 + j];
#pragma unroll
      for (int j = 0; j < 8; ++j) t[8 + j] = lds[widx][lr][b0 + 16 + j];
      v16bf a = pack_a16(t);
#pragma unroll
      for (int nt = 0; nt < 8; ++nt) {
        v16us bu = *(const v16us*)(W2p + ((size_t)(kt * 8 + nt)) * 512 + (size_t)lane * 16);
        acc2[nt] = wmma_bf16(a, __builtin_bit_cast(v16bf, bu), acc2[nt]);
      }
    }
#pragma unroll
    for (int nt = 0; nt < 8; ++nt) {
      int ocol = nt * 16 + lr;
      float bv = b2[ocol];
#pragma unroll
      for (int r = 0; r < 8; ++r) {
        int orow = wave * 16 + hi * 8 + r;
        if (orow < NP1) {
          size_t off = (size_t)orow * HID + ocol;
          xout[off] = acc2[nt][r] + bv + x[off];   // residual
        }
      }
    }
  }
}

__global__ void __launch_bounds__(256) copy_out_kernel(const float4* __restrict__ src,
                                                       float4* __restrict__ dst, int n4) {
  int i = blockIdx.x * blockDim.x + threadIdx.x;
  if (i < n4) dst[i] = src[i];
}

// ---------------------------------------------------------------------------
extern "C" void kernel_launch(void* const* d_in, const int* in_sizes, int n_in,
                              void* d_out, int out_size, void* d_ws, size_t ws_size,
                              hipStream_t stream) {
  const float* pep      = (const float*)d_in[0];
  const int*   ei       = (const int*)d_in[1];
  const int*   ea       = (const int*)d_in[2];
  const float* Wproj    = (const float*)d_in[3];
  const float* bproj    = (const float*)d_in[4];
  const float* emb      = (const float*)d_in[5];
  const float* cls_tok  = (const float*)d_in[6];
  const float* cls_edge = (const float*)d_in[7];
  const float* W1a = (const float*)d_in[8];
  const float* b1a = (const float*)d_in[9];
  const float* W2a = (const float*)d_in[10];
  const float* b2a = (const float*)d_in[11];
  const float* W1b = (const float*)d_in[12];
  const float* b1b = (const float*)d_in[13];
  const float* W2b = (const float*)d_in[14];
  const float* b2b = (const float*)d_in[15];

  const size_t X_BYTES = (size_t)NP1 * HID * sizeof(float);   // 51,200,512
  char* ws = (char*)d_ws;
  float* x   = (float*)(ws);
  float* agg = (float*)(ws + X_BYTES);
  unsigned short* wp  = (unsigned short*)(ws + 2 * X_BYTES);
  unsigned short* w1a = wp  + 65536;   // Wproj packed: 16 ktiles * 8 ntiles * 512
  unsigned short* w2a = w1a + 16384;   // 4 * 8 * 512 each
  unsigned short* w1b = w2a + 16384;
  unsigned short* w2b = w1b + 16384;

  // 1) pack weights to WMMA-ready bf16 B layout
  pack_w_kernel<<<128, 32, 0, stream>>>(Wproj, wp);
  pack_w_kernel<<<32, 32, 0, stream>>>(W1a, w1a);
  pack_w_kernel<<<32, 32, 0, stream>>>(W2a, w2a);
  pack_w_kernel<<<32, 32, 0, stream>>>(W1b, w1b);
  pack_w_kernel<<<32, 32, 0, stream>>>(W2b, w2b);

  // 2) x = pep @ Wproj + b ;  x[N] = cls_token
  proj_gemm_kernel<<<(N_NODES / 16 + 3) / 4, 128, 0, stream>>>(pep, wp, bproj, x);
  cls_init_kernel<<<1, 128, 0, stream>>>(cls_tok, x);

  const int n4_agg = NP1 * (HID / 4);
  const int edge_blocks = (int)(((long long)(N_EDGES + N_NODES) * (HID / 4) + 255) / 256);
  const int mlp_blocks = ((NP1 + 15) / 16 + 3) / 4;

  // 3) layer A
  zero_kernel<<<(n4_agg + 255) / 256, 256, 0, stream>>>((float4*)agg, n4_agg);
  edge_msg_kernel<<<edge_blocks, 256, 0, stream>>>(x, ei, ea, emb, cls_edge, agg);
  mlp_kernel<<<mlp_blocks, 128, 0, stream>>>(x, agg, w1a, b1a, w2a, b2a, x);

  // 4) layer B
  zero_kernel<<<(n4_agg + 255) / 256, 256, 0, stream>>>((float4*)agg, n4_agg);
  edge_msg_kernel<<<edge_blocks, 256, 0, stream>>>(x, ei, ea, emb, cls_edge, agg);
  mlp_kernel<<<mlp_blocks, 128, 0, stream>>>(x, agg, w1b, b1b, w2b, b2b, x);

  // 5) d_out = x[0:N]
  const int n4_out = N_NODES * (HID / 4);
  copy_out_kernel<<<(n4_out + 255) / 256, 256, 0, stream>>>((const float4*)x, (float4*)d_out, n4_out);
}
